// GraphAttnModule_60524679135346
// MI455X (gfx1250) — matrix operand order repaired and checked by reading, not compiled, not run on hardware
//
#include <hip/hip_runtime.h>
#include <hip/hip_bf16.h>

// ---------------------------------------------------------------------------
// Problem constants (from reference): B=4, S=256, N=32, H=768, heads=12, dh=64
//   BS = 1024, M_neib = BS*N = 32768, M_node = 1024, K = H = 768
//   n-tiles = 768/16 = 48, k-tiles = 768/32 = 24
// ---------------------------------------------------------------------------
#define H_DIM     768
#define N_NEIB    32
#define BS_TOT    1024
#define M_NEIB    32768
#define M_NODE    1024
#define NTILES    48
#define KTILES    24
#define NEG_VAL   (-10000000000.0f)

typedef __attribute__((ext_vector_type(16))) __bf16 v16bf;
typedef __attribute__((ext_vector_type(8)))  __bf16 v8bf;
typedef __attribute__((ext_vector_type(4)))  __bf16 v4bf;
typedef __attribute__((ext_vector_type(8)))  float  v8f;

// ---------------------------------------------------------------------------
// Stage 1: pack W (f32 [768,768], row = output feature o, col = input k) into
// the WMMA bf16 B-operand layout. B is 32x16 (KxN) per k-tile.
//   Per-lane 16 bf16 values stored contiguously (32B):
//     lane %16 -> column n;  lane<16 -> K = kt*32 + 0..15
//                            lane>=16 -> K = kt*32 + 16..31
//   Wpk element index = ((nt*KTILES + kt)*32 + lane)*16 + e
// ---------------------------------------------------------------------------
__global__ void pack_w_kernel(const float* __restrict__ W, __bf16* __restrict__ Wpk) {
    int idx = blockIdx.x * 256 + threadIdx.x;
    if (idx >= NTILES * KTILES * 32 * 16) return;
    int e    = idx & 15;
    int lane = (idx >> 4) & 31;
    int kt   = (idx >> 9) % KTILES;
    int nt   = (idx >> 9) / KTILES;
    int n = nt * 16 + (lane & 15);
    int k = kt * 32 + ((lane < 16) ? 0 : 16) + e;
    Wpk[idx] = (__bf16)W[n * H_DIM + k];   // B[k][n] = W[n][k]  (y = x @ W^T)
}

// ---------------------------------------------------------------------------
// Stage 2: f32 -> bf16 row-major conversion of activations (done once so the
// GEMM inner loop is pure loads + WMMA, and A-traffic is halved).
// ---------------------------------------------------------------------------
__global__ void cvt_bf16_kernel(const float* __restrict__ src,
                                __bf16* __restrict__ dst, int n4) {
    int i = blockIdx.x * 256 + threadIdx.x;
    if (i >= n4) return;
    float4 v = ((const float4*)src)[i];
    v4bf o;
    o[0] = (__bf16)v.x; o[1] = (__bf16)v.y; o[2] = (__bf16)v.z; o[3] = (__bf16)v.w;
    *(v4bf*)(dst + 4 * i) = o;
}

// ---------------------------------------------------------------------------
// Stage 3: Y[M,768] = X[M,768] @ W^T via v_wmma_f32_16x16x32_bf16.
// Block = 256 threads = 8 waves. blockIdx.x = m-tile (16 rows).
// Wave w computes n-tiles [6w, 6w+6)  -> 48 f32 accumulator VGPRs per wave.
//
// Address plan: one per-lane base pointer for A and one for B; all loads in
// the k-loop use compile-time immediate offsets (B n-tile stride = 24576 B,
// fits imm24) and the k-step advances both pointers by a constant.
//
// Scheduling: sched_group_barrier hints force the 14 VMEM reads of an
// iteration to issue as a front-loaded batch before the 6 WMMAs, so the wait
// inserter can use partial s_wait_loadcnt values and the load clause of the
// next iteration overlaps the WMMA burst of the current one.
//
// A per-lane layout (contiguous k, same order as packed B): row m = mTile*16 + lane%16,
//   lane<16: K = kt*32 + 0..15;  lane>=16: +16..31
// ---------------------------------------------------------------------------
__global__ __launch_bounds__(256) void gemm_wmma_bf16_kernel(
    const __bf16* __restrict__ X,    // [M, 768] bf16 row-major
    const __bf16* __restrict__ Wpk,  // packed B operand tiles
    float* __restrict__ Y)           // [M, 768] f32
{
    const int lane   = threadIdx.x & 31;
    const int wave   = threadIdx.x >> 5;
    const int mTile  = blockIdx.x;
    const int laneLo = lane & 15;
    const int hi     = (lane >= 16) ? 1 : 0;

    // Per-lane bases (all k-loop addressing is immediate offsets from these).
    const __bf16* aptr = X + (size_t)(mTile * 16 + laneLo) * H_DIM + (hi ? 16 : 0);
    const __bf16* bptr = Wpk + ((size_t)(wave * 6 * KTILES) * 32 + lane) * 16;

    v8f acc[6];
#pragma unroll
    for (int j = 0; j < 6; ++j)
#pragma unroll
        for (int i = 0; i < 8; ++i) acc[j][i] = 0.0f;

    for (int kt = 0; kt < KTILES; ++kt) {
        // ---- issue the whole load batch first ----
        v8bf alo = *(const v8bf*)(aptr);          // K k0..k0+7
        v8bf ahi = *(const v8bf*)(aptr + 8);      // K k0+8..k0+15
        v16bf b[6];
#pragma unroll
        for (int j = 0; j < 6; ++j)               // imm offset: j*24576 bytes
            b[j] = *(const v16bf*)(bptr + (size_t)j * (KTILES * 512));

        v16bf a;
#pragma unroll
        for (int e = 0; e < 8; ++e) { a[e] = alo[e]; a[8 + e] = ahi[e]; }

        // ---- 6 WMMAs against the shared A fragment ----
#pragma unroll
        for (int j = 0; j < 6; ++j)
            acc[j] = __builtin_amdgcn_wmma_f32_16x16x32_bf16(
                false, a, false, b[j], (short)0, acc[j], false, false);

#if __has_builtin(__builtin_amdgcn_sched_group_barrier)
        // Desired per-iteration schedule: 14 VMEM reads, then 6 WMMA.
        __builtin_amdgcn_sched_group_barrier(0x020, 14, 0);  // VMEM read group
        __builtin_amdgcn_sched_group_barrier(0x008, 6, 0);   // WMMA group
#endif

        aptr += 32;    // next k-tile in the activation row (64 bytes)
        bptr += 512;   // next k-tile block in packed W (1 KB)
    }

    // D layout: VGPR i -> row M = i + (hi?8:0), col N = lane%16
#pragma unroll
    for (int j = 0; j < 6; ++j) {
        const int col = (wave * 6 + j) * 16 + laneLo;
        float* yb = Y + (size_t)(mTile * 16 + (hi ? 8 : 0)) * H_DIM + col;
#pragma unroll
        for (int i = 0; i < 8; ++i) yb[(size_t)i * H_DIM] = acc[j][i];
    }
}

// ---------------------------------------------------------------------------
// Stage 4: fused attention. One block (256 threads) per (b,s).
// Heads 0..5 score from node projection (broadcast over n); heads 6..11 score
// from neighbor projection slice (h-6)*128. LeakyReLU(0.01) -> mask -> softmax
// over N=32 -> out[h,d] = sum_n prob * neibP[n, h*64+d].
// ---------------------------------------------------------------------------
__global__ __launch_bounds__(256) void attn_kernel(
    const float* __restrict__ nodeP,   // [1024, 768]
    const float* __restrict__ neibP,   // [1024*32, 768]
    const int*   __restrict__ mask,    // [1024*32]
    const float* __restrict__ attnM,   // [12*128]
    float* __restrict__ out)           // [1024, 768]
{
    __shared__ float sAttn[12 * 128];
    __shared__ float sNode[H_DIM];
    __shared__ float sScore[12 * N_NEIB];
    __shared__ float sMax[12], sSum[12];

    const int bs  = blockIdx.x;
    const int tid = threadIdx.x;
    const float* nb = neibP + (size_t)bs * N_NEIB * H_DIM;

    for (int i = tid; i < 12 * 128; i += 256) sAttn[i] = attnM[i];
    for (int i = tid; i < H_DIM;    i += 256) sNode[i] = nodeP[(size_t)bs * H_DIM + i];
    __syncthreads();

    // scores: 384 (h,n) pairs; h is wave-uniform per iteration (p>>5)
    for (int p = tid; p < 12 * N_NEIB; p += 256) {
        const int h = p >> 5, n = p & 31;
        const float* ap = sAttn + h * 128;
        float s = 0.0f;
        if (h < 6) {
            const float* xp = sNode + h * 128;
#pragma unroll 4
            for (int d = 0; d < 128; ++d) s += xp[d] * ap[d];
        } else {
            const float* xp = nb + (size_t)n * H_DIM + (h - 6) * 128;
#pragma unroll 4
            for (int d = 0; d < 128; ++d) s += xp[d] * ap[d];
        }
        s = (s >= 0.0f) ? s : 0.01f * s;                // LeakyReLU(0.01)
        if (mask[bs * N_NEIB + n] == 0) s = NEG_VAL;    // mask after activation
        sScore[p] = s;
    }
    __syncthreads();

    if (tid < 12) {
        float m = sScore[tid * 32];
        for (int n = 1; n < 32; ++n) m = fmaxf(m, sScore[tid * 32 + n]);
        float s = 0.0f;
        for (int n = 0; n < 32; ++n) s += __expf(sScore[tid * 32 + n] - m);
        sMax[tid] = m; sSum[tid] = s;
    }
    __syncthreads();

    for (int p = tid; p < 12 * N_NEIB; p += 256) {
        const int h = p >> 5;
        sScore[p] = __expf(sScore[p] - sMax[h]) / sSum[h];
    }
    __syncthreads();

    for (int o = tid; o < H_DIM; o += 256) {
        const int h = o >> 6, d = o & 63;
        const float* base = nb + h * 64 + d;
        const float* pr = sScore + h * 32;
        float s = 0.0f;
#pragma unroll
        for (int n = 0; n < 32; ++n) s += pr[n] * base[(size_t)n * H_DIM];
        out[(size_t)bs * H_DIM + o] = s;
    }
}

// ---------------------------------------------------------------------------
// Launch. Workspace layout (bytes):
//   [0)               Wpk      bf16  1,179,648
//   [1179648)         node_proj f32  3,145,728
//   [4325376)         neib_proj f32  100,663,296
//   [104988672)       node_bf16      1,572,864
//   [106561536)       neib_bf16      50,331,648   (total ~149.6 MB)
// ---------------------------------------------------------------------------
extern "C" void kernel_launch(void* const* d_in, const int* in_sizes, int n_in,
                              void* d_out, int out_size, void* d_ws, size_t ws_size,
                              hipStream_t stream) {
    const float* node  = (const float*)d_in[0];   // [4,256,768]
    const float* neib  = (const float*)d_in[1];   // [4,256,32,768]
    const int*   mask  = (const int*)  d_in[2];   // [4,256,32]
    const float* W     = (const float*)d_in[3];   // [768,768]
    const float* attnM = (const float*)d_in[4];   // [12,128,1]
    float*       out   = (float*)d_out;           // [4,256,768]

    char* ws = (char*)d_ws;
    __bf16* Wpk   = (__bf16*)(ws);
    float*  nodeP = (float*) (ws + 1179648);
    float*  neibP = (float*) (ws + 4325376);
    __bf16* nodeB = (__bf16*)(ws + 104988672);
    __bf16* neibB = (__bf16*)(ws + 106561536);

    // 1. pack W into WMMA B-operand layout (bf16)
    pack_w_kernel<<<(NTILES * KTILES * 32 * 16 + 255) / 256, 256, 0, stream>>>(W, Wpk);

    // 2. convert activations to bf16 (float4-granular)
    cvt_bf16_kernel<<<((M_NODE * H_DIM / 4) + 255) / 256, 256, 0, stream>>>(
        node, nodeB, M_NODE * H_DIM / 4);
    cvt_bf16_kernel<<<((M_NEIB * H_DIM / 4) + 255) / 256, 256, 0, stream>>>(
        neib, neibB, M_NEIB * H_DIM / 4);

    // 3. projections via bf16 WMMA (f32 accumulate)
    gemm_wmma_bf16_kernel<<<M_NODE / 16, 256, 0, stream>>>(nodeB, Wpk, nodeP);
    gemm_wmma_bf16_kernel<<<M_NEIB / 16, 256, 0, stream>>>(neibB, Wpk, neibP);

    // 4. fused attention (scores -> leaky -> mask -> softmax -> weighted sum)
    attn_kernel<<<BS_TOT, 256, 0, stream>>>(nodeP, neibP, mask, attnM, out);
}
